// IVRAG_78520592105863
// MI455X (gfx1250) — matrix operand order repaired and checked by reading, not compiled, not run on hardware
//
#include <hip/hip_runtime.h>

// ---------------------------------------------------------------------------
// Types for CDNA5 WMMA / TDM
// ---------------------------------------------------------------------------
typedef __bf16 bf16_t;
typedef __attribute__((ext_vector_type(16))) __bf16 v16bf_t;
typedef __attribute__((ext_vector_type(8)))  __bf16 v8bf_t;
typedef __attribute__((ext_vector_type(8)))  float  v8f_t;
typedef __attribute__((ext_vector_type(4))) unsigned int u32x4_t;
typedef __attribute__((ext_vector_type(8))) int i32x8_t;
typedef __attribute__((ext_vector_type(4))) int i32x4_t;

#define B_SZ    1024
#define CONF    64
#define TREAT   2
#define EMB     512
#define TOPK    8
#define HID     1024
#define INSTR   20
#define NDOCS   200000

#define NCHUNK  128                    // docs per workgroup in sim kernel
#define NCHUNKS 1563                   // ceil(200000/128)
#define KSTEPS  (EMB / 32)             // 16

// fs_in: [instr(20) | flat(4096)] = 4116 -> pad 4128
#define FSK 4128
// ss_in: [treat(2) | conf(64) | flat(4096)] = 4162 -> pad 4192
#define SSK 4192

// d_out element offsets (f32 elements)
#define OUT_OUTCOME 0        // 1024
#define OUT_SCORES  1024     // 8192
#define OUT_INDICES 9216     // 8192 (int32 bits)
#define OUT_PRED    17408    // 2048
#define OUT_IV      19456    // 2048

#if __has_builtin(__builtin_amdgcn_tensor_load_to_lds)
#define HAVE_TDM 1
#endif

__device__ __forceinline__ void top8_insert(float* s, int* id, float v, int idx) {
    if (v <= s[7]) return;
    s[7] = v; id[7] = idx;
#pragma unroll
    for (int j = 6; j >= 0; --j) {
        if (v > s[j]) {
            float ts = s[j]; int ti = id[j];
            s[j] = v; id[j] = idx;
            s[j + 1] = ts; id[j + 1] = ti;
        }
    }
}

// ---------------------------------------------------------------------------
// A-tile (16x512 bf16 = 16 KB) prefetch into LDS.
// Preferred: Tensor Data Mover, one flat 1-D descriptor (8192 x 2B elements),
// issued once by wave 0, tracked by TENSORcnt.
// Fallback: per-lane global_load_async_to_lds_b128 (4 async instrs per wave).
// ---------------------------------------------------------------------------
#ifdef HAVE_TDM
__device__ __forceinline__ void tdm_load_a_tile(const bf16_t* src, unsigned lds_addr) {
    const unsigned long long ga = (unsigned long long)(uintptr_t)src;
    u32x4_t g0;
    g0[0] = 1u;                                    // count=1, user descriptor
    g0[1] = lds_addr;                              // D#.lds_addr
    g0[2] = (unsigned)(ga & 0xffffffffu);          // global_addr[31:0]
    g0[3] = (unsigned)((ga >> 32) & 0x01ffffffu)   // global_addr[56:32]
          | (2u << 30);                            // type = 2 ("image")
    i32x8_t g1;
    g1[0] = (int)(1u << 16);                       // data_size=1 (2 bytes), no multicast
    g1[1] = (int)((8192u & 0xffffu) << 16);        // tensor_dim0[15:0] = 8192
    g1[2] = (int)((8192u >> 16) | (1u << 16));     // tensor_dim0[31:16] | tensor_dim1[15:0]=1
    g1[3] = (int)(8192u << 16);                    // tensor_dim1[31:16]=0 | tile_dim0 = 8192
    g1[4] = (int)1u;                               // tile_dim1 = 1, tile_dim2 = 0
    g1[5] = (int)8192u;                            // tensor_dim0_stride[31:0] = 8192
    g1[6] = 0;                                     // stride0 hi | stride1 lo
    g1[7] = 0;                                     // stride1 hi
    i32x4_t g2 = {0, 0, 0, 0};
    i32x4_t g3 = {0, 0, 0, 0};
    i32x8_t g4 = {0, 0, 0, 0, 0, 0, 0, 0};        // extra group (clang-23 6-arg form)
    __builtin_amdgcn_tensor_load_to_lds(g0, g1, g2, g3, g4, 0);
}
#define TILE_WAIT_ONE(tid)  do { if ((tid) < 32) __builtin_amdgcn_s_wait_tensorcnt(1); } while (0)
#define TILE_WAIT_ZERO(tid) do { if ((tid) < 32) __builtin_amdgcn_s_wait_tensorcnt(0); } while (0)
#else
#define TILE_WAIT_ONE(tid)  asm volatile("s_wait_asynccnt 0x4" ::: "memory")
#define TILE_WAIT_ZERO(tid) asm volatile("s_wait_asynccnt 0x0" ::: "memory")
#endif

__device__ __forceinline__ void issue_a_tile(const bf16_t* __restrict__ pembB, int mt,
                                             bf16_t* dst, int tid) {
#ifdef HAVE_TDM
    if (tid < 32)
        tdm_load_a_tile(pembB + (size_t)mt * 16 * EMB, (unsigned)(uintptr_t)dst);
#else
    const bf16_t* src = pembB + (size_t)mt * 16 * EMB;
#pragma unroll
    for (int i = 0; i < 4; ++i) {                   // 1024 b128 transfers / 256 threads
        const int e = tid + i * 256;
        const unsigned lo = (unsigned)(uintptr_t)dst + (unsigned)e * 16u;
        const unsigned long long ga = (unsigned long long)(uintptr_t)src
                                    + (unsigned long long)e * 16u;
        asm volatile("global_load_async_to_lds_b128 %0, %1, off"
                     :: "v"(lo), "v"(ga) : "memory");
    }
#endif
}

// ---------------------------------------------------------------------------
// patient_emb = l2norm(patient @ W_enc.T + b_enc) -> bf16 [1024][512]
// ---------------------------------------------------------------------------
__launch_bounds__(256)
__global__ void k_patient_emb(const float* __restrict__ patient,
                              const float* __restrict__ W_enc,
                              const float* __restrict__ b_enc,
                              bf16_t* __restrict__ pembB) {
    __shared__ float prow[CONF + TREAT];
    __shared__ float emb[EMB];
    __shared__ float red[8];
    const int p = blockIdx.x, tid = threadIdx.x;
    if (tid < CONF + TREAT) prow[tid] = patient[(size_t)p * (CONF + TREAT) + tid];
    __syncthreads();
    float ss = 0.f;
    for (int o = tid; o < EMB; o += 256) {
        const float* w = W_enc + (size_t)o * (CONF + TREAT);
        float acc = b_enc[o];
#pragma unroll 11
        for (int k = 0; k < CONF + TREAT; ++k) acc += prow[k] * w[k];
        emb[o] = acc;
        ss += acc * acc;
    }
#pragma unroll
    for (int m = 16; m >= 1; m >>= 1) ss += __shfl_xor(ss, m, 32);
    if ((tid & 31) == 0) red[tid >> 5] = ss;
    __syncthreads();
    float tot = 0.f;
#pragma unroll
    for (int i = 0; i < 8; ++i) tot += red[i];
    const float rn = 1.f / fmaxf(sqrtf(tot), 1e-12f);
    for (int o = tid; o < EMB; o += 256)
        pembB[(size_t)p * EMB + o] = (bf16_t)(emb[o] * rn);
}

// ---------------------------------------------------------------------------
// corpus reciprocal row norms: 1 wave per row
// ---------------------------------------------------------------------------
__launch_bounds__(256)
__global__ void k_corpus_rnorm(const float* __restrict__ corpus,
                               float* __restrict__ rnorm) {
    const int row = blockIdx.x * 8 + (threadIdx.x >> 5);
    const int lane = threadIdx.x & 31;
    if (row >= NDOCS) return;
    const float4* p = (const float4*)(corpus + (size_t)row * EMB);
    float ss = 0.f;
#pragma unroll
    for (int i = 0; i < 4; ++i) {
        float4 v = p[lane + i * 32];
        ss += v.x * v.x + v.y * v.y + v.z * v.z + v.w * v.w;
    }
#pragma unroll
    for (int m = 16; m >= 1; m >>= 1) ss += __shfl_xor(ss, m, 32);
    if (lane == 0) rnorm[row] = 1.f / fmaxf(sqrtf(ss), 1e-12f);
}

// ---------------------------------------------------------------------------
// Fused similarity GEMM (bf16 WMMA) + per-chunk top-8.
// Workgroup = 128-doc chunk resident in LDS (normalized bf16). A-tiles are
// double-buffered and prefetched by the Tensor Data Mover (overlapped with
// WMMA on the current tile). Each wave owns 16 docs.
// ---------------------------------------------------------------------------
__launch_bounds__(256)
__global__ void k_sim_topk(const float* __restrict__ corpus,
                           const float* __restrict__ rnorm,
                           const bf16_t* __restrict__ pembB,
                           float* __restrict__ cand_s,
                           int* __restrict__ cand_i) {
    __shared__ bf16_t ldsB[NCHUNK][EMB];     // 128 KB: normalized corpus chunk
    __shared__ bf16_t ldsA[2][16][EMB];      // 32 KB : double-buffered A tiles
    __shared__ float  ldsS[16][NCHUNK];      // 8 KB  : scores for one M-tile

    const int blk = blockIdx.x, tid = threadIdx.x;
    const long n_start = (long)blk * NCHUNK;
    const int n_count = (int)min((long)NCHUNK, (long)NDOCS - n_start);

    // prefetch A tile 0 (TDM) before doing the heavy corpus staging
    issue_a_tile(pembB, 0, &ldsA[0][0][0], tid);

    // ---- load + normalize + convert corpus chunk into LDS (zero-pad tail) ----
#pragma unroll 4
    for (int it = 0; it < (NCHUNK * EMB) / (256 * 4); ++it) {
        const int e = (it * 256 + tid) * 4;
        const int d = e / EMB, k = e % EMB;
        float4 v = make_float4(0.f, 0.f, 0.f, 0.f);
        if (d < n_count) {
            v = *(const float4*)(corpus + (n_start + d) * EMB + k);
            const float r = rnorm[n_start + d];
            v.x *= r; v.y *= r; v.z *= r; v.w *= r;
        }
        bf16_t* q = &ldsB[d][k];
        q[0] = (bf16_t)v.x; q[1] = (bf16_t)v.y; q[2] = (bf16_t)v.z; q[3] = (bf16_t)v.w;
    }

    const int wave = tid >> 5, lane = tid & 31;
    const int laneM = lane & 15, laneH = lane >> 4;
    const int nw0 = wave * 16;       // this wave's 16-doc column tile

    for (int mt = 0; mt < B_SZ / 16; ++mt) {
        // prefetch next A tile into the other buffer, then wait for current
        if (mt + 1 < B_SZ / 16) {
            issue_a_tile(pembB, mt + 1, &ldsA[(mt + 1) & 1][0][0], tid);
            TILE_WAIT_ONE(tid);
        } else {
            TILE_WAIT_ZERO(tid);
        }
        __syncthreads();             // publish A tile (and corpus on iter 0)

        const bf16_t (*lA)[EMB] = ldsA[mt & 1];
        v8f_t c = {};
#pragma unroll
        for (int ks = 0; ks < KSTEPS; ++ks) {
            const int k0 = ks * 32;
            // A frag (16x32): lanes 0-15 row M=lane, K {0..15}; lanes 16-31 K {16..31}
            const v8bf_t alo = *(const v8bf_t*)&lA[laneM][k0 + laneH * 16];
            const v8bf_t ahi = *(const v8bf_t*)&lA[laneM][k0 + laneH * 16 + 8];
            const v16bf_t a = __builtin_shufflevector(alo, ahi,
                0, 1, 2, 3, 4, 5, 6, 7, 8, 9, 10, 11, 12, 13, 14, 15);
            // B frag (32x16): lane = doc column; 16 contiguous K per lane
            const v16bf_t b = *(const v16bf_t*)&ldsB[nw0 + laneM][k0 + laneH * 16];
            c = __builtin_amdgcn_wmma_f32_16x16x32_bf16(false, a, false, b,
                                                        (short)0, c, false, false);
        }
        // C layout: vgpr r, lanes 0-15 -> (M=r, N=lane); lanes 16-31 -> (M=r+8, N=lane-16)
#pragma unroll
        for (int r = 0; r < 8; ++r)
            ldsS[r + laneH * 8][nw0 + laneM] = c[r];
        __syncthreads();

        if (tid < 16) {
            float s[8]; int id[8];
#pragma unroll
            for (int j = 0; j < 8; ++j) { s[j] = -__builtin_inff(); id[j] = 0; }
            for (int n = 0; n < n_count; ++n)
                top8_insert(s, id, ldsS[tid][n], (int)(n_start + n));
            const size_t base = ((size_t)(mt * 16 + tid) * NCHUNKS + blk) * 8;
#pragma unroll
            for (int j = 0; j < 8; ++j) { cand_s[base + j] = s[j]; cand_i[base + j] = id[j]; }
        }
        __syncthreads();
    }
}

// ---------------------------------------------------------------------------
// Final top-8 reduce (1563*8 candidates -> 8) + gather retrieved rows into
// bf16 MLP input buffers (concat handled by column offsets) + pads.
// ---------------------------------------------------------------------------
__launch_bounds__(256)
__global__ void k_final_topk_gather(const float* __restrict__ cand_s,
                                    const int* __restrict__ cand_i,
                                    const float* __restrict__ corpus,
                                    const float* __restrict__ instruments,
                                    const float* __restrict__ confounders,
                                    float* __restrict__ out,
                                    bf16_t* __restrict__ fs_in,
                                    bf16_t* __restrict__ ss_in) {
    __shared__ float ls[256 * 8];
    __shared__ int   li[256 * 8];
    __shared__ int   docs[TOPK];
    const int p = blockIdx.x, tid = threadIdx.x;
    const int NC = NCHUNKS * 8;

    float s[8]; int id[8];
#pragma unroll
    for (int j = 0; j < 8; ++j) { s[j] = -__builtin_inff(); id[j] = 0; }
    const size_t base = (size_t)p * NC;
    for (int i = tid; i < NC; i += 256)
        top8_insert(s, id, cand_s[base + i], cand_i[base + i]);
#pragma unroll
    for (int j = 0; j < 8; ++j) { ls[tid * 8 + j] = s[j]; li[tid * 8 + j] = id[j]; }
    __syncthreads();
    if (tid == 0) {
        float fs[8]; int fi[8];
#pragma unroll
        for (int j = 0; j < 8; ++j) { fs[j] = -__builtin_inff(); fi[j] = 0; }
        for (int i = 0; i < 256 * 8; ++i) top8_insert(fs, fi, ls[i], li[i]);
        int* oi = (int*)out;
#pragma unroll
        for (int j = 0; j < 8; ++j) {
            out[OUT_SCORES + p * 8 + j] = fs[j];
            oi[OUT_INDICES + p * 8 + j] = fi[j];
            docs[j] = fi[j];
        }
    }
    __syncthreads();
    for (int j = tid; j < TOPK * EMB; j += 256) {
        const int k = j >> 9, e = j & 511;
        const bf16_t bv = (bf16_t)corpus[(size_t)docs[k] * EMB + e];
        fs_in[(size_t)p * FSK + INSTR + j] = bv;
        ss_in[(size_t)p * SSK + TREAT + CONF + j] = bv;
    }
    if (tid < INSTR)
        fs_in[(size_t)p * FSK + tid] = (bf16_t)instruments[(size_t)p * INSTR + tid];
    if (tid >= INSTR && tid < INSTR + (FSK - (INSTR + TOPK * EMB)))
        fs_in[(size_t)p * FSK + INSTR + TOPK * EMB + (tid - INSTR)] = (bf16_t)0.f;
    if (tid < CONF)
        ss_in[(size_t)p * SSK + TREAT + tid] = (bf16_t)confounders[(size_t)p * CONF + tid];
    if (tid >= CONF && tid < CONF + (SSK - (TREAT + CONF + TOPK * EMB)))
        ss_in[(size_t)p * SSK + TREAT + CONF + TOPK * EMB + (tid - CONF)] = (bf16_t)0.f;
    // ss_in cols [0,2) written by the L1b GEMM epilogue
}

// ---------------------------------------------------------------------------
// Weight convert f32 [N][K] -> zero-padded bf16 [Npad][Kpad]
// ---------------------------------------------------------------------------
__launch_bounds__(256)
__global__ void k_wcvt(const float* __restrict__ W, int N, int K,
                       bf16_t* __restrict__ Wb, int Npad, int Kpad) {
    const size_t total = (size_t)Npad * Kpad;
    const size_t stride = (size_t)gridDim.x * blockDim.x;
    for (size_t i = (size_t)blockIdx.x * blockDim.x + threadIdx.x; i < total; i += stride) {
        const int n = (int)(i / Kpad), k = (int)(i % Kpad);
        const float v = (n < N && k < K) ? W[(size_t)n * K + k] : 0.f;
        Wb[i] = (bf16_t)v;
    }
}

// ---------------------------------------------------------------------------
// Generic bf16 WMMA GEMM: out = act(A[M][Ka] @ Wb[Npad][Ka]^T + bias)
// wave computes 32x32 (2x2 WMMA tiles); block 256 thr = 128x64 tile.
// ---------------------------------------------------------------------------
__launch_bounds__(256)
__global__ void k_gemm(const bf16_t* __restrict__ A, int Ka,
                       const bf16_t* __restrict__ Wb, int Npad,
                       const float* __restrict__ bias, int Nreal, int relu,
                       float* outF, int outFstride, int outFcol,
                       bf16_t* outB, int outBstride, int outBcol) {
    const int wave = threadIdx.x >> 5, lane = threadIdx.x & 31;
    const int laneM = lane & 15, laneH = lane >> 4;
    const int m0 = blockIdx.x * 128 + (wave >> 1) * 32;
    const int n0 = blockIdx.y * 64 + (wave & 1) * 32;
    if (n0 >= Npad) return;

    v8f_t c00 = {}, c01 = {}, c10 = {}, c11 = {};
    const bf16_t* arow0 = A + (size_t)(m0 + laneM) * Ka + laneH * 16;
    const bf16_t* arow1 = A + (size_t)(m0 + 16 + laneM) * Ka + laneH * 16;
    const bf16_t* wrow0 = Wb + (size_t)(n0 + laneM) * Ka + laneH * 16;
    const bf16_t* wrow1 = Wb + (size_t)(n0 + 16 + laneM) * Ka + laneH * 16;

    for (int k0 = 0; k0 < Ka; k0 += 32) {
        const v8bf_t a0lo = *(const v8bf_t*)(arow0 + k0);
        const v8bf_t a0hi = *(const v8bf_t*)(arow0 + k0 + 8);
        const v8bf_t a1lo = *(const v8bf_t*)(arow1 + k0);
        const v8bf_t a1hi = *(const v8bf_t*)(arow1 + k0 + 8);
        const v16bf_t a0 = __builtin_shufflevector(a0lo, a0hi,
            0, 1, 2, 3, 4, 5, 6, 7, 8, 9, 10, 11, 12, 13, 14, 15);
        const v16bf_t a1 = __builtin_shufflevector(a1lo, a1hi,
            0, 1, 2, 3, 4, 5, 6, 7, 8, 9, 10, 11, 12, 13, 14, 15);
        const v16bf_t b0 = *(const v16bf_t*)(wrow0 + k0);
        const v16bf_t b1 = *(const v16bf_t*)(wrow1 + k0);
        c00 = __builtin_amdgcn_wmma_f32_16x16x32_bf16(false, a0, false, b0, (short)0, c00, false, false);
        c01 = __builtin_amdgcn_wmma_f32_16x16x32_bf16(false, a0, false, b1, (short)0, c01, false, false);
        c10 = __builtin_amdgcn_wmma_f32_16x16x32_bf16(false, a1, false, b0, (short)0, c10, false, false);
        c11 = __builtin_amdgcn_wmma_f32_16x16x32_bf16(false, a1, false, b1, (short)0, c11, false, false);
    }

#pragma unroll
    for (int t = 0; t < 4; ++t) {
        const v8f_t c = (t == 0) ? c00 : (t == 1) ? c01 : (t == 2) ? c10 : c11;
        const int mB = m0 + ((t >> 1) ? 16 : 0);
        const int n = n0 + ((t & 1) ? 16 : 0) + laneM;
        if (n >= Nreal) continue;
        const float bv = bias[n];
#pragma unroll
        for (int r = 0; r < 8; ++r) {
            const int m = mB + r + laneH * 8;
            float v = c[r] + bv;
            if (relu) v = fmaxf(v, 0.f);
            if (outF) outF[(size_t)m * outFstride + outFcol + n] = v;
            if (outB) outB[(size_t)m * outBstride + outBcol + n] = (bf16_t)v;
        }
    }
}

// ---------------------------------------------------------------------------
// iv_strength = instruments @ W_iv.T + b_iv   (tiny, VALU)
// ---------------------------------------------------------------------------
__launch_bounds__(256)
__global__ void k_iv(const float* __restrict__ instr,
                     const float* __restrict__ W_iv,
                     const float* __restrict__ b_iv,
                     float* __restrict__ out) {
    const int i = blockIdx.x * blockDim.x + threadIdx.x;
    if (i >= B_SZ * TREAT) return;
    const int p = i >> 1, t = i & 1;
    float acc = b_iv[t];
#pragma unroll
    for (int k = 0; k < INSTR; ++k)
        acc += instr[(size_t)p * INSTR + k] * W_iv[(size_t)t * INSTR + k];
    out[OUT_IV + i] = acc;
}

// ---------------------------------------------------------------------------
// Host launch
// ---------------------------------------------------------------------------
static inline size_t alignUp(size_t x, size_t a) { return (x + a - 1) & ~(a - 1); }

extern "C" void kernel_launch(void* const* d_in, const int* in_sizes, int n_in,
                              void* d_out, int out_size, void* d_ws, size_t ws_size,
                              hipStream_t stream) {
    (void)in_sizes; (void)n_in; (void)out_size; (void)ws_size;
    const float* patient     = (const float*)d_in[0];
    const float* confounders = (const float*)d_in[1];
    const float* instruments = (const float*)d_in[2];
    const float* corpus      = (const float*)d_in[3];
    const float* W_enc = (const float*)d_in[4];
    const float* b_enc = (const float*)d_in[5];
    const float* W1a = (const float*)d_in[6];
    const float* b1a = (const float*)d_in[7];
    const float* W1b = (const float*)d_in[8];
    const float* b1b = (const float*)d_in[9];
    const float* W2a = (const float*)d_in[10];
    const float* b2a = (const float*)d_in[11];
    const float* W2b = (const float*)d_in[12];
    const float* b2b = (const float*)d_in[13];
    const float* W2c = (const float*)d_in[14];
    const float* b2c = (const float*)d_in[15];
    const float* W_iv = (const float*)d_in[16];
    const float* b_iv = (const float*)d_in[17];
    float* out = (float*)d_out;

    char* ws = (char*)d_ws;
    size_t off = 0;
    auto alloc = [&](size_t bytes) { size_t o = off; off = alignUp(off + bytes, 256); return o; };
    float*  rnorm  = (float*)(ws + alloc((size_t)NDOCS * 4));
    bf16_t* pembB  = (bf16_t*)(ws + alloc((size_t)B_SZ * EMB * 2));
    float*  cand_s = (float*)(ws + alloc((size_t)B_SZ * NCHUNKS * 8 * 4));
    int*    cand_i = (int*)(ws + alloc((size_t)B_SZ * NCHUNKS * 8 * 4));
    bf16_t* fs_in  = (bf16_t*)(ws + alloc((size_t)B_SZ * FSK * 2));
    bf16_t* ss_in  = (bf16_t*)(ws + alloc((size_t)B_SZ * SSK * 2));
    bf16_t* h_a    = (bf16_t*)(ws + alloc((size_t)B_SZ * HID * 2));
    bf16_t* h_b    = (bf16_t*)(ws + alloc((size_t)B_SZ * (HID / 2) * 2));
    bf16_t* W1a_b  = (bf16_t*)(ws + alloc((size_t)HID * FSK * 2));
    bf16_t* W1b_b  = (bf16_t*)(ws + alloc((size_t)32 * HID * 2));
    bf16_t* W2a_b  = (bf16_t*)(ws + alloc((size_t)HID * SSK * 2));
    bf16_t* W2b_b  = (bf16_t*)(ws + alloc((size_t)(HID / 2) * HID * 2));
    bf16_t* W2c_b  = (bf16_t*)(ws + alloc((size_t)32 * (HID / 2) * 2));

    // prep
    k_patient_emb<<<B_SZ, 256, 0, stream>>>(patient, W_enc, b_enc, pembB);
    k_corpus_rnorm<<<NDOCS / 8, 256, 0, stream>>>(corpus, rnorm);
    k_wcvt<<<2048, 256, 0, stream>>>(W1a, HID, INSTR + TOPK * EMB, W1a_b, HID, FSK);
    k_wcvt<<<128, 256, 0, stream>>>(W1b, TREAT, HID, W1b_b, 32, HID);
    k_wcvt<<<2048, 256, 0, stream>>>(W2a, HID, TREAT + CONF + TOPK * EMB, W2a_b, HID, SSK);
    k_wcvt<<<1024, 256, 0, stream>>>(W2b, HID / 2, HID, W2b_b, HID / 2, HID);
    k_wcvt<<<64, 256, 0, stream>>>(W2c, 1, HID / 2, W2c_b, 32, HID / 2);

    // retrieval: fused WMMA GEMM + chunk top-8, then global reduce + gather
    k_sim_topk<<<NCHUNKS, 256, 0, stream>>>(corpus, rnorm, pembB, cand_s, cand_i);
    k_final_topk_gather<<<B_SZ, 256, 0, stream>>>(cand_s, cand_i, corpus,
                                                  instruments, confounders,
                                                  out, fs_in, ss_in);

    // MLPs (WMMA)
    k_gemm<<<dim3(B_SZ / 128, HID / 64), 256, 0, stream>>>(
        fs_in, FSK, W1a_b, HID, b1a, HID, 1,
        nullptr, 0, 0, h_a, HID, 0);
    k_gemm<<<dim3(B_SZ / 128, 1), 256, 0, stream>>>(
        h_a, HID, W1b_b, 32, b1b, TREAT, 0,
        out + OUT_PRED, TREAT, 0, ss_in, SSK, 0);
    k_gemm<<<dim3(B_SZ / 128, HID / 64), 256, 0, stream>>>(
        ss_in, SSK, W2a_b, HID, b2a, HID, 1,
        nullptr, 0, 0, h_a, HID, 0);
    k_gemm<<<dim3(B_SZ / 128, (HID / 2) / 64), 256, 0, stream>>>(
        h_a, HID, W2b_b, HID / 2, b2b, HID / 2, 1,
        nullptr, 0, 0, h_b, HID / 2, 0);
    k_gemm<<<dim3(B_SZ / 128, 1), 256, 0, stream>>>(
        h_b, HID / 2, W2c_b, 32, b2c, 1, 0,
        out + OUT_OUTCOME, 1, 0, nullptr, 0, 0);

    k_iv<<<(B_SZ * TREAT + 255) / 256, 256, 0, stream>>>(instruments, W_iv, b_iv, out);
}
